// GCNZinc_47777216201364
// MI455X (gfx1250) — matrix operand order repaired and checked by reading, not compiled, not run on hardware
//
#include <hip/hip_runtime.h>
#include <hip/hip_bf16.h>
#include <stdint.h>

#define N_NODES 30000
#define N_EDGES 480000
#define EMB_DIM 512
#define NUM_REL 4

typedef __attribute__((ext_vector_type(16))) __bf16 v16bf;
typedef __attribute__((ext_vector_type(2)))  __bf16 v2bf;
typedef __attribute__((ext_vector_type(8)))  float  v8f;
typedef __attribute__((ext_vector_type(4)))  float  v4f;
typedef __attribute__((ext_vector_type(4)))  int    v4i;

// global (addrspace 1) qualified load pointer: guarantees global_load_b128,
// never flat_load (flat would tie up DScnt and serialize against LDS ops).
// NOTE: must point at a clang ext-vector type (trivial), not HIP's float4 class.
typedef __attribute__((address_space(1))) const v4f* gv4f_p;

// ---------- f32x2 -> packed bf16x2 (RNE) ----------
#if __has_builtin(__builtin_amdgcn_cvt_pk_bf16_f32)
__device__ __forceinline__ uint32_t pack2_bf16(float a, float b) {
    v2bf r = __builtin_amdgcn_cvt_pk_bf16_f32(a, b);
    return __builtin_bit_cast(uint32_t, r);
}
#else
__device__ __forceinline__ uint32_t pack2_bf16(float a, float b) {
    uint32_t ua = __builtin_bit_cast(uint32_t, a);
    uint32_t ub = __builtin_bit_cast(uint32_t, b);
    ua += 0x7FFFu + ((ua >> 16) & 1u);
    ub += 0x7FFFu + ((ub >> 16) & 1u);
    return (ua >> 16) | (ub & 0xFFFF0000u);
}
#endif

// ---------- async global->LDS b128 copy (CDNA5), with safe fallback ----------
#if __has_builtin(__builtin_amdgcn_global_load_async_to_lds_b128)
#define HAVE_ASYNC_LDS 1
typedef __attribute__((address_space(1))) v4i* g_v4i_p;
typedef __attribute__((address_space(3))) v4i* l_v4i_p;
__device__ __forceinline__ void async_cp_b128(const void* gsrc, void* ldst) {
    __builtin_amdgcn_global_load_async_to_lds_b128((g_v4i_p)gsrc, (l_v4i_p)ldst, 0, 0);
}
__device__ __forceinline__ void async_wait0() {
#if __has_builtin(__builtin_amdgcn_s_wait_asynccnt)
    __builtin_amdgcn_s_wait_asynccnt(0);
#else
    asm volatile("s_wait_asynccnt 0x0" ::: "memory");
#endif
}
#else
#define HAVE_ASYNC_LDS 0
__device__ __forceinline__ void async_cp_b128(const void* gsrc, void* ldst) {
    *(uint4*)ldst = *(const uint4*)gsrc;
}
__device__ __forceinline__ void async_wait0() {}
#endif

// ---------- zero fill ----------
__global__ void zero_f32(float4* __restrict__ p, long n4) {
    long i = (long)blockIdx.x * blockDim.x + threadIdx.x;
    long stride = (long)gridDim.x * blockDim.x;
    float4 z = make_float4(0.f, 0.f, 0.f, 0.f);
    for (; i < n4; i += stride) p[i] = z;
}

// ---------- x = emb[x_idx] ----------
__global__ void embed_k(const int* __restrict__ idx,
                        const float4* __restrict__ emb,
                        float4* __restrict__ x) {
    long i = (long)blockIdx.x * blockDim.x + threadIdx.x;
    long total = (long)N_NODES * (EMB_DIM / 4);
    if (i >= total) return;
    int n = (int)(i >> 7);
    int c = (int)(i & 127);
    x[(long)n * 128 + c] = emb[(long)idx[n] * 128 + c];
}

// ---------- edge scatter: msg[r][dst] += x[src]; cnt[r][dst] += 1 ----------
__global__ void scatter_k(const float4* __restrict__ x,
                          const int* __restrict__ ei,     // [2, E]
                          const int* __restrict__ et,     // [E]
                          float* __restrict__ msg,
                          float* __restrict__ cnt) {
    long gid = (long)blockIdx.x * blockDim.x + threadIdx.x;
    if (gid >= (long)N_EDGES * 128) return;
    int e = (int)(gid >> 7);
    int c = (int)(gid & 127);
    int s = ei[e];
    int d = ei[N_EDGES + e];
    int r = et[e];
    float4 v = x[(long)s * 128 + c];
    float* o = msg + ((size_t)r * N_NODES + d) * EMB_DIM + c * 4;
    unsafeAtomicAdd(o + 0, v.x);
    unsafeAtomicAdd(o + 1, v.y);
    unsafeAtomicAdd(o + 2, v.z);
    unsafeAtomicAdd(o + 3, v.w);
    if (c == 0) unsafeAtomicAdd(cnt + (size_t)r * N_NODES + d, 1.0f);
}

// ---------- cnt -> 1/max(cnt,1) ----------
__global__ void invcnt_k(float* __restrict__ c) {
    int i = blockIdx.x * blockDim.x + threadIdx.x;
    if (i < NUM_REL * N_NODES) c[i] = 1.0f / fmaxf(c[i], 1.0f);
}

// ---------- pack [W(4x512x512) ; root(512x512)] -> bf16 ----------
__global__ void convw_k(const float* __restrict__ W,
                        const float* __restrict__ root,
                        uint16_t* __restrict__ wb) {
    int i = blockIdx.x * blockDim.x + threadIdx.x;
    const int wtot = NUM_REL * EMB_DIM * EMB_DIM;
    const int tot = wtot + EMB_DIM * EMB_DIM;
    if (i >= tot) return;
    float f = (i < wtot) ? W[i] : root[i - wtot];
    uint32_t u = __builtin_bit_cast(uint32_t, f);
    u += 0x7FFFu + ((u >> 16) & 1u);
    wb[i] = (uint16_t)(u >> 16);
}

// ---------- fused RGCN GEMM ----------
// block: 256 threads (8 waves). Output tile 128(M) x 128(N), double-buffered LDS.
// wave (wm 0..3, wn 0..1) owns a 32x64 sub-tile = 2x4 wmma accumulators.
#define A_STRIDE 80           // (32+8) bf16 per row, bytes
#define B_STRIDE 272          // (128+8) bf16 per row, bytes
#define A_BYTES (128 * A_STRIDE)
#define B_BYTES (32 * B_STRIDE)
#define N_ITERS 80            // 5 slabs * 16 k-steps

__global__ __launch_bounds__(256) void rgcn_gemm_k(
        const float* __restrict__ msg,      // [4, N, 512] f32
        const float* __restrict__ invc,     // [4, N] f32
        const float* __restrict__ xin,      // [N, 512] f32
        const uint16_t* __restrict__ wb,    // [5*512, 512] bf16
        const float* __restrict__ bias,     // [512] f32
        float* __restrict__ xout)           // [N, 512] f32
{
    __shared__ __align__(16) uint8_t ldsA[2 * A_BYTES];
    __shared__ __align__(16) uint8_t ldsB[2 * B_BYTES];

    const int t = threadIdx.x;
    const int lane = t & 31;
    const int wave = t >> 5;
    const int wm = wave & 3;
    const int wn = wave >> 2;
    const int block_m = blockIdx.y * 128;
    const int block_n = blockIdx.x * 128;

    v8f acc[2][4];
#pragma unroll
    for (int a = 0; a < 2; ++a)
#pragma unroll
        for (int b = 0; b < 4; ++b) {
            v8f z = {0.f, 0.f, 0.f, 0.f, 0.f, 0.f, 0.f, 0.f};
            acc[a][b] = z;
        }

    // staging roles
    const int rowA  = t >> 1;          // 0..127
    const int halfA = t & 1;           // k half
    const int kB    = t >> 3;          // 0..31
    const int pB    = t & 7;           // n chunk of 16
    const int mrowA = block_m + rowA;
    const bool inb  = (mrowA < N_NODES);

    // Two base pointers + uniform slab arithmetic (r = it>>4 is wave-uniform,
    // so slab selection is one scalar-condition select, and loads stay global).
    const size_t SLAB = (size_t)N_NODES * EMB_DIM;   // floats per relation slab
    const float* mbase = msg + (size_t)mrowA * EMB_DIM + halfA * 16;
    const float* xbase = xin + (size_t)mrowA * EMB_DIM + halfA * 16;

    float ascl[5];
#pragma unroll
    for (int r = 0; r < 4; ++r)
        ascl[r] = inb ? invc[(size_t)r * N_NODES + mrowA] : 0.0f;
    ascl[4] = 1.0f;

    const uint16_t* bbase = wb + (size_t)kB * EMB_DIM + block_n + pB * 16;

    // per-lane fragment offsets (A layout: m = lane&15, kb = (lane>>4)*8)
    const int fa_row = wm * 32 + (lane & 15);
    const int fa_kb  = (lane >> 4) * 8;

    // ---- staging of iteration `it` into buffer `buf` ----
    auto stage = [&](int it, int buf) {
        const int r  = it >> 4;
        const int kk = it & 15;
        // A: 16 f32 -> bf16 (scaled); forced addrspace(1) loads
        uint4 a0, a1;
        if (inb) {
            const float scl = ascl[r];
            const float* src = (r < 4) ? (mbase + SLAB * (size_t)r) : xbase;
            gv4f_p g = (gv4f_p)(src + (size_t)kk * 32);
            v4f f0 = g[0], f1 = g[1], f2 = g[2], f3 = g[3];
            a0.x = pack2_bf16(f0.x * scl, f0.y * scl);
            a0.y = pack2_bf16(f0.z * scl, f0.w * scl);
            a0.z = pack2_bf16(f1.x * scl, f1.y * scl);
            a0.w = pack2_bf16(f1.z * scl, f1.w * scl);
            a1.x = pack2_bf16(f2.x * scl, f2.y * scl);
            a1.y = pack2_bf16(f2.z * scl, f2.w * scl);
            a1.z = pack2_bf16(f3.x * scl, f3.y * scl);
            a1.w = pack2_bf16(f3.z * scl, f3.w * scl);
        } else {
            a0 = make_uint4(0u, 0u, 0u, 0u);
            a1 = a0;
        }
        uint8_t* abuf = &ldsA[buf * A_BYTES + rowA * A_STRIDE + halfA * 32];
        *(uint4*)(abuf)      = a0;
        *(uint4*)(abuf + 16) = a1;
        // B: raw bf16 copy (async global->LDS on CDNA5)
        const uint8_t* gsrc =
            (const uint8_t*)(bbase + ((size_t)r * EMB_DIM + (size_t)kk * 32) * EMB_DIM);
        uint8_t* bbuf = &ldsB[buf * B_BYTES + kB * B_STRIDE + pB * 32];
        async_cp_b128(gsrc,      bbuf);
        async_cp_b128(gsrc + 16, bbuf + 16);
    };

    // ---- prologue ----
    stage(0, 0);
    async_wait0();
    __syncthreads();

    for (int it = 0; it < N_ITERS; ++it) {
        const int cur = it & 1;
        if (it + 1 < N_ITERS) stage(it + 1, (it + 1) & 1);

        union Frag { uint4 u[2]; v16bf v; };
        Frag af[2], bf[4];
        const uint8_t* Ab = &ldsA[cur * A_BYTES];
        const uint8_t* Bb = &ldsB[cur * B_BYTES];
#pragma unroll
        for (int tm = 0; tm < 2; ++tm) {
            int ro = (fa_row + tm * 16) * A_STRIDE + fa_kb * 2;
            af[tm].u[0] = *(const uint4*)&Ab[ro];
            af[tm].u[1] = *(const uint4*)&Ab[ro + 32];
        }
#pragma unroll
        for (int tn = 0; tn < 4; ++tn) {
            int co = (wn * 64 + tn * 16 + (lane & 15)) * 2;
            int kb = (lane >> 4) * 8;
#pragma unroll
            for (int s = 0; s < 16; ++s) {
                int k = (s < 8) ? (kb + s) : (16 + kb + s - 8);
                bf[tn].v[s] = __builtin_bit_cast(__bf16, *(const uint16_t*)&Bb[k * B_STRIDE + co]);
            }
        }
#pragma unroll
        for (int tm = 0; tm < 2; ++tm)
#pragma unroll
            for (int tn = 0; tn < 4; ++tn)
                acc[tm][tn] = __builtin_amdgcn_wmma_f32_16x16x32_bf16(
                    false, af[tm].v, false, bf[tn].v,
                    (short)0, acc[tm][tn], false, false);

        async_wait0();       // staged async B-tile landed in LDS
        __syncthreads();     // everyone done reading cur + writing next
    }

    // ---- epilogue: bias + relu + residual ----
    const int nidx = lane & 15;
    const int mo   = (lane >> 4) * 8;
#pragma unroll
    for (int tm = 0; tm < 2; ++tm) {
        int rb = block_m + wm * 32 + tm * 16 + mo;
#pragma unroll
        for (int tn = 0; tn < 4; ++tn) {
            int n = block_n + wn * 64 + tn * 16 + nidx;
            float bn = bias[n];
#pragma unroll
            for (int i = 0; i < 8; ++i) {
                int m = rb + i;
                if (m < N_NODES) {
                    float v = acc[tm][tn][i] + bn;
                    v = fmaxf(v, 0.0f) + xin[(size_t)m * EMB_DIM + n];
                    xout[(size_t)m * EMB_DIM + n] = v;
                }
            }
        }
    }
}

// ---------- host ----------
extern "C" void kernel_launch(void* const* d_in, const int* in_sizes, int n_in,
                              void* d_out, int out_size, void* d_ws, size_t ws_size,
                              hipStream_t stream) {
    const int*   x_idx = (const int*)d_in[0];
    const int*   ei    = (const int*)d_in[1];
    const int*   et    = (const int*)d_in[2];
    const float* emb   = (const float*)d_in[3];
    const float* Ws[3] = {(const float*)d_in[4], (const float*)d_in[7], (const float*)d_in[10]};
    const float* Rs[3] = {(const float*)d_in[5], (const float*)d_in[8], (const float*)d_in[11]};
    const float* Bs[3] = {(const float*)d_in[6], (const float*)d_in[9], (const float*)d_in[12]};

    char* ws = (char*)d_ws;
    float*    xA  = (float*)(ws);                     //  61,440,000 B
    float*    msg = (float*)(ws + 61440000);          // 245,760,000 B
    float*    cnt = (float*)(ws + 307200000);         //     480,000 B
    uint16_t* wbf = (uint16_t*)(ws + 307680000);      //   2,621,440 B

    {
        long total = (long)N_NODES * (EMB_DIM / 4);
        embed_k<<<dim3((unsigned)((total + 255) / 256)), dim3(256), 0, stream>>>(
            x_idx, (const float4*)emb, (float4*)xA);
    }

    const float* xin = xA;
    float* xout = (float*)d_out;

    for (int l = 0; l < 3; ++l) {
        long msg4 = (long)NUM_REL * N_NODES * EMB_DIM / 4;
        zero_f32<<<dim3(4096), dim3(256), 0, stream>>>((float4*)msg, msg4);
        zero_f32<<<dim3(128), dim3(256), 0, stream>>>((float4*)cnt,
                                                      (long)NUM_REL * N_NODES / 4);

        long sc = (long)N_EDGES * 128;
        scatter_k<<<dim3((unsigned)((sc + 255) / 256)), dim3(256), 0, stream>>>(
            (const float4*)xin, ei, et, msg, cnt);

        invcnt_k<<<dim3((NUM_REL * N_NODES + 255) / 256), dim3(256), 0, stream>>>(cnt);

        convw_k<<<dim3((5 * EMB_DIM * EMB_DIM + 255) / 256), dim3(256), 0, stream>>>(
            Ws[l], Rs[l], wbf);

        dim3 grid(EMB_DIM / 128, (N_NODES + 127) / 128);   // (4, 235), n-tiles fastest
        rgcn_gemm_k<<<grid, dim3(256), 0, stream>>>(msg, cnt, xin, wbf, Bs[l], xout);

        const float* tmp = xin;
        xin  = xout;
        xout = (float*)tmp;
    }
}
